// Trans_Encoder_30880814858668
// MI455X (gfx1250) — matrix operand order repaired and checked by reading, not compiled, not run on hardware
//
#include <hip/hip_runtime.h>
#include <hip/hip_bf16.h>

// Problem constants (match reference)
constexpr int Bc = 16;    // batch
constexpr int Sc = 2048;  // seq len
constexpr int Kc = 256;   // knowledge dim
constexpr int Dc = 512;   // embed dim

typedef __attribute__((ext_vector_type(16))) _Float16 v16h;
typedef __attribute__((ext_vector_type(8)))  _Float16 v8h;
typedef __attribute__((ext_vector_type(8)))  float    v8f;

__device__ __forceinline__ v16h pack16(v8h lo, v8h hi) {
  v16h r;
#pragma unroll
  for (int i = 0; i < 8; ++i) { r[i] = lo[i]; r[8 + i] = hi[i]; }
  return r;
}

// A-operand tile 16(M) x 32(K), f16, source row-major with leading dim lda.
// ISA layout: lane m=lane&15, half=lane>>4; VGPR0-3 hold K=8*half..8*half+7,
// VGPR4-7 hold K=16+8*half..16+8*half+7  -> two contiguous 16B chunks.
__device__ __forceinline__ v16h load_a16(const _Float16* base, int lda, int lane) {
  int m = lane & 15, half = lane >> 4;
  const _Float16* p = base + (size_t)m * lda + 8 * half;
  v8h lo = *(const v8h*)(p);
  v8h hi = *(const v8h*)(p + 16);
  return pack16(lo, hi);
}

// B-operand tile 32(K) x 16(N), f16, source row-major [K,N] leading dim ldb.
// ISA layout: lane n=lane&15, half=lane>>4; elems 0-7 hold K=8*half..8*half+7,
// elems 8-15 hold K=16+8*half..16+8*half+7 (strided by ldb).
__device__ __forceinline__ v16h load_b16(const _Float16* base, int ldb, int lane) {
  int n = lane & 15, half = lane >> 4;
  const _Float16* p = base + (size_t)(8 * half) * ldb + n;
  v16h r;
#pragma unroll
  for (int i = 0; i < 8; ++i) {
    r[i] = p[(size_t)i * ldb];
    r[8 + i] = p[(size_t)(16 + i) * ldb];
  }
  return r;
}

__device__ __forceinline__ v8f wmma_f16(v16h a, v16h b, v8f c) {
  return __builtin_amdgcn_wmma_f32_16x16x32_f16(false, a, false, b, (short)0, c,
                                                false, false);
}

// ---------------- small utility kernels ----------------

__global__ void zero_f32(float* p, int n) {
  int i = blockIdx.x * blockDim.x + threadIdx.x;
  if (i < n) p[i] = 0.0f;
}

__global__ void cvt_f32_f16(const float* in, _Float16* out, int n) {
  int i = blockIdx.x * blockDim.x + threadIdx.x;
  if (i < n) out[i] = (_Float16)in[i];
}

// x[bs,k] = emb_table[data[bs]][k] * know[bs,k]   (f16 result)
__global__ void gather_mul(const int* __restrict__ data,
                           const float* __restrict__ emb,
                           const float* __restrict__ know,
                           _Float16* __restrict__ x) {
  int bs = blockIdx.x;
  int row = data[bs];
  for (int t = threadIdx.x; t < Kc; t += blockDim.x) {
    x[(size_t)bs * Kc + t] =
        (_Float16)(emb[(size_t)row * Kc + t] * know[(size_t)bs * Kc + t]);
  }
}

// kT[b][d][s] = ksrc[b][s][d]
__global__ void transpose_f16(const _Float16* __restrict__ src,
                              _Float16* __restrict__ dst) {
  __shared__ _Float16 tile[32][33];
  int b = blockIdx.z;
  int s0 = blockIdx.x * 32, d0 = blockIdx.y * 32;
  int tx = threadIdx.x & 31, ty = threadIdx.x >> 5;  // 32 x 8
#pragma unroll
  for (int i = 0; i < 32; i += 8)
    tile[ty + i][tx] = src[(size_t)(b * Sc + s0 + ty + i) * Dc + d0 + tx];
  __syncthreads();
#pragma unroll
  for (int i = 0; i < 32; i += 8)
    dst[(size_t)b * Dc * Sc + (size_t)(d0 + ty + i) * Sc + s0 + tx] =
        tile[tx][ty + i];
}

// ---------------- WMMA GEMM:  C[M,N](f16) = A[M,Kd](f16) @ W[Kd,N](f16) + bias(f32)
// block = 256 threads = 8 waves; each wave owns one 32x64 output strip
// (2 M-tiles x 4 N-tiles) so every B tile feeds two WMMAs. Kd is a template
// constant (256 or 512) so the k-loop fully unrolls: the scheduler pipelines
// A/B loads across steps without register-rotation movs or hazard nops.
template <int Kd>
__global__ __launch_bounds__(256) void gemm_f16_nn(
    const _Float16* __restrict__ A, const _Float16* __restrict__ W,
    const float* __restrict__ bias, _Float16* __restrict__ C, int M, int N) {
  int wave = threadIdx.x >> 5, lane = threadIdx.x & 31;
  int nJobN = N >> 6;  // strips of 64 along N
  int job = blockIdx.x * 8 + wave;
  int tm = job / nJobN, tng = job % nJobN;
  int row0 = tm * 32, n0 = tng * 64;
  if (row0 >= M) return;

  const _Float16* Abase = A + (size_t)row0 * Kd;
  v8f acc0[4] = {}, acc1[4] = {};

#pragma unroll
  for (int kk = 0; kk < Kd; kk += 32) {
    v16h a0 = load_a16(Abase + kk, Kd, lane);
    v16h a1 = load_a16(Abase + (size_t)16 * Kd + kk, Kd, lane);
#pragma unroll
    for (int t = 0; t < 4; ++t) {
      v16h b = load_b16(W + (size_t)kk * N + n0 + 16 * t, N, lane);
      acc0[t] = wmma_f16(a0, b, acc0[t]);
      acc1[t] = wmma_f16(a1, b, acc1[t]);
    }
  }
  int n = lane & 15, half = lane >> 4;
#pragma unroll
  for (int t = 0; t < 4; ++t) {
    float bv = bias[n0 + 16 * t + n];
#pragma unroll
    for (int r = 0; r < 8; ++r) {
      float v0 = acc0[t][r] + bv;
      float v1 = acc1[t][r] + bv;
      C[(size_t)(row0 + r + 8 * half) * N + n0 + 16 * t + n] = (_Float16)v0;
      C[(size_t)(row0 + 16 + r + 8 * half) * N + n0 + 16 * t + n] = (_Float16)v1;
    }
  }
}

// ---------------- Flash attention + query-sum ----------------
// grid = (S/16, B); block = 256 (8 waves). Wave w owns d-chunk [64w, 64w+64).
// Produces sa[b][d] += sum over the 16 queries of softmax(qK^T/sqrt(D)) V.
__global__ __launch_bounds__(256) void attn_kernel(
    const _Float16* __restrict__ q, const _Float16* __restrict__ kT,
    const _Float16* __restrict__ v, const int* __restrict__ lens,
    float* __restrict__ sa) {
  __shared__ float part[8][2][16][16];   // per-wave partial scores
  __shared__ _Float16 pshare[16][32];    // softmaxed P tile (f16)
  __shared__ float alpha_s[16];          // per-row rescale
  __shared__ float linv_s[16];           // 1/row-sum

  int w = threadIdx.x >> 5, lane = threadIdx.x & 31;
  int col = lane & 15, half = lane >> 4;
  int b = blockIdx.y;
  int q0 = blockIdx.x * 16;
  int d0 = w * 64;
  int len = lens[b];  // key s valid iff s <= len
  const float scale = rsqrtf((float)Dc);

  const _Float16* qbase = q + (size_t)(b * Sc + q0) * Dc;
  v16h qa0 = load_a16(qbase + d0, Dc, lane);
  v16h qa1 = load_a16(qbase + d0 + 32, Dc, lane);

  v8f o[4] = {};
  float mrun = -1e30f, lrun = 0.0f;  // meaningful in wave0 lanes 0..15

  for (int kb = 0; kb < Sc; kb += 32) {
    if (kb > len) break;  // uniform: remaining key tiles fully masked

    // (1) partial scores over this wave's 64-wide d-chunk
    v8f s0 = {}, s1 = {};
    {
      const _Float16* ktb = kT + (size_t)b * Dc * Sc + (size_t)d0 * Sc + kb;
      v16h b0 = load_b16(ktb, Sc, lane);
      v16h b1 = load_b16(ktb + 16, Sc, lane);
      s0 = wmma_f16(qa0, b0, s0);
      s1 = wmma_f16(qa0, b1, s1);
      ktb += (size_t)32 * Sc;
      b0 = load_b16(ktb, Sc, lane);
      b1 = load_b16(ktb + 16, Sc, lane);
      s0 = wmma_f16(qa1, b0, s0);
      s1 = wmma_f16(qa1, b1, s1);
    }
    __syncthreads();  // previous iteration done reading pshare
#pragma unroll
    for (int r = 0; r < 8; ++r) {
      part[w][0][r + 8 * half][col] = s0[r];
      part[w][1][r + 8 * half][col] = s1[r];
    }
    __syncthreads();

    // (2) wave0: cross-wave reduce + streaming softmax, one row per lane
    if (w == 0 && lane < 16) {
      int row = lane;
      float sv[32];
      float tmax = -1e30f;
#pragma unroll
      for (int j = 0; j < 32; ++j) {
        float acc = 0.0f;
#pragma unroll
        for (int ww = 0; ww < 8; ++ww) acc += part[ww][j >> 4][row][j & 15];
        acc *= scale;
        acc = ((kb + j) <= len) ? acc : -1e30f;
        sv[j] = acc;
        tmax = fmaxf(tmax, acc);
      }
      float mnew = fmaxf(mrun, tmax);
      float al = __expf(mrun - mnew);
      float ps = 0.0f;
#pragma unroll
      for (int j = 0; j < 32; ++j) {
        float p = __expf(sv[j] - mnew);
        pshare[row][j] = (_Float16)p;
        ps += p;
      }
      lrun = lrun * al + ps;
      mrun = mnew;
      alpha_s[row] = al;
    }
    __syncthreads();

    // (3) O = alpha*O + P @ V_tile  on this wave's d-chunk
    float alf[8];
#pragma unroll
    for (int r = 0; r < 8; ++r) alf[r] = alpha_s[r + 8 * half];
    v16h pa = load_a16(&pshare[0][0], 32, lane);
#pragma unroll
    for (int t = 0; t < 4; ++t) {
#pragma unroll
      for (int r = 0; r < 8; ++r) o[t][r] *= alf[r];
      const _Float16* vb = v + (size_t)(b * Sc + kb) * Dc + d0 + 16 * t;
      v16h bv = load_b16(vb, Dc, lane);
      o[t] = wmma_f16(pa, bv, o[t]);
    }
  }

  // epilogue: sum_q O[q][d] / l[q]  -> atomic into sa[b][d]
  if (w == 0 && lane < 16) linv_s[lane] = 1.0f / lrun;
  __syncthreads();
  float rinv[8];
#pragma unroll
  for (int r = 0; r < 8; ++r) rinv[r] = linv_s[r + 8 * half];
#pragma unroll
  for (int t = 0; t < 4; ++t) {
    float acc = 0.0f;
#pragma unroll
    for (int r = 0; r < 8; ++r) acc += o[t][r] * rinv[r];
    acc += __shfl_xor(acc, 16, 32);
    if (half == 0) atomicAdd(&sa[(size_t)b * Dc + d0 + 16 * t + col], acc);
  }
}

// ---------------- tiny epilogue ----------------
// h[b][st*D + d] = sa[st][b] @ Wo_st + S*bo_st   (sum over S queries folds bias)
__global__ void head_proj(const float* __restrict__ sa,
                          const float* __restrict__ Wo0,
                          const float* __restrict__ bo0,
                          const float* __restrict__ Wo1,
                          const float* __restrict__ bo1,
                          float* __restrict__ h) {
  int idx = blockIdx.x * blockDim.x + threadIdx.x;  // B*2*D
  if (idx >= Bc * 2 * Dc) return;
  int d = idx % Dc;
  int rest = idx / Dc;
  int st = rest & 1;
  int b = rest >> 1;
  const float* W = st ? Wo1 : Wo0;
  const float* bo = st ? bo1 : bo0;
  const float* s = sa + (size_t)st * Bc * Dc + (size_t)b * Dc;
  float acc = bo[d] * (float)Sc;
  for (int k2 = 0; k2 < Dc; ++k2) acc += s[k2] * W[(size_t)k2 * Dc + d];
  h[(size_t)b * 2 * Dc + st * Dc + d] = acc;
}

__global__ void out_proj(const float* __restrict__ h,
                         const float* __restrict__ Wout,
                         const float* __restrict__ bout,
                         float* __restrict__ out) {
  int idx = blockIdx.x * blockDim.x + threadIdx.x;  // B*K
  if (idx >= Bc * Kc) return;
  int n = idx % Kc, b = idx / Kc;
  float acc = bout[n];
  for (int j = 0; j < 2 * Dc; ++j)
    acc += h[(size_t)b * 2 * Dc + j] * Wout[(size_t)j * Kc + n];
  out[idx] = tanhf(acc);
}

// ---------------- host-side orchestration ----------------

extern "C" void kernel_launch(void* const* d_in, const int* in_sizes, int n_in,
                              void* d_out, int out_size, void* d_ws,
                              size_t ws_size, hipStream_t stream) {
  (void)in_sizes; (void)n_in; (void)out_size; (void)ws_size;
  const int*   data[2] = {(const int*)d_in[0],  (const int*)d_in[3]};
  const float* know[2] = {(const float*)d_in[1], (const float*)d_in[4]};
  const int*   lens[2] = {(const int*)d_in[2],  (const int*)d_in[5]};
  const float* embT = (const float*)d_in[6];
  const float* Wt   = (const float*)d_in[7];
  const float* bt   = (const float*)d_in[8];
  const float* Wq[2] = {(const float*)d_in[9],  (const float*)d_in[17]};
  const float* bq[2] = {(const float*)d_in[10], (const float*)d_in[18]};
  const float* Wk[2] = {(const float*)d_in[11], (const float*)d_in[19]};
  const float* bk[2] = {(const float*)d_in[12], (const float*)d_in[20]};
  const float* Wv[2] = {(const float*)d_in[13], (const float*)d_in[21]};
  const float* bv[2] = {(const float*)d_in[14], (const float*)d_in[22]};
  const float* Wo[2] = {(const float*)d_in[15], (const float*)d_in[23]};
  const float* bo[2] = {(const float*)d_in[16], (const float*)d_in[24]};
  const float* Wout = (const float*)d_in[25];
  const float* bout = (const float*)d_in[26];
  float* out = (float*)d_out;

  // workspace carve-up (~176 MB)
  const size_t BS = (size_t)Bc * Sc;
  size_t off = 0;
  auto carve = [&](size_t bytes) {
    size_t o = off;
    off = (off + bytes + 255) & ~(size_t)255;
    return o;
  };
  char* ws = (char*)d_ws;
  _Float16* xf16  = (_Float16*)(ws + carve(BS * Kc * 2));
  _Float16* emb16 = (_Float16*)(ws + carve(BS * Dc * 2));
  _Float16* q16   = (_Float16*)(ws + carve(BS * Dc * 2));
  _Float16* k16   = (_Float16*)(ws + carve(BS * Dc * 2));
  _Float16* kT16  = (_Float16*)(ws + carve(BS * Dc * 2));
  _Float16* v16   = (_Float16*)(ws + carve(BS * Dc * 2));
  _Float16* wt16  = (_Float16*)(ws + carve((size_t)Kc * Dc * 2));
  _Float16* wq16  = (_Float16*)(ws + carve((size_t)Dc * Dc * 2));
  _Float16* wk16  = (_Float16*)(ws + carve((size_t)Dc * Dc * 2));
  _Float16* wv16  = (_Float16*)(ws + carve((size_t)Dc * Dc * 2));
  float*    sa    = (float*)(ws + carve((size_t)2 * Bc * Dc * 4));
  float*    h     = (float*)(ws + carve((size_t)Bc * 2 * Dc * 4));

  zero_f32<<<(2 * Bc * Dc + 255) / 256, 256, 0, stream>>>(sa, 2 * Bc * Dc);

  const int M = (int)BS;                                  // 32768
  const int gemmBlocks = (M / 32) * (Dc / 64) / 8;        // 1024
  dim3 trGrid(Sc / 32, Dc / 32, Bc);
  dim3 attnGrid(Sc / 16, Bc);

  for (int st = 0; st < 2; ++st) {
    cvt_f32_f16<<<(Kc * Dc + 255) / 256, 256, 0, stream>>>(Wt, wt16, Kc * Dc);
    cvt_f32_f16<<<(Dc * Dc + 255) / 256, 256, 0, stream>>>(Wq[st], wq16, Dc * Dc);
    cvt_f32_f16<<<(Dc * Dc + 255) / 256, 256, 0, stream>>>(Wk[st], wk16, Dc * Dc);
    cvt_f32_f16<<<(Dc * Dc + 255) / 256, 256, 0, stream>>>(Wv[st], wv16, Dc * Dc);

    gather_mul<<<M, 256, 0, stream>>>(data[st], embT, know[st], xf16);

    gemm_f16_nn<Kc><<<gemmBlocks, 256, 0, stream>>>(xf16, wt16, bt, emb16, M, Dc);
    gemm_f16_nn<Dc><<<gemmBlocks, 256, 0, stream>>>(emb16, wq16, bq[st], q16, M, Dc);
    gemm_f16_nn<Dc><<<gemmBlocks, 256, 0, stream>>>(emb16, wk16, bk[st], k16, M, Dc);
    gemm_f16_nn<Dc><<<gemmBlocks, 256, 0, stream>>>(emb16, wv16, bv[st], v16, M, Dc);

    transpose_f16<<<trGrid, 256, 0, stream>>>(k16, kT16);

    attn_kernel<<<attnGrid, 256, 0, stream>>>(q16, kT16, v16, lens[st],
                                              sa + (size_t)st * Bc * Dc);
  }

  head_proj<<<(Bc * 2 * Dc + 255) / 256, 256, 0, stream>>>(sa, Wo[0], bo[0],
                                                           Wo[1], bo[1], h);
  out_proj<<<(Bc * Kc + 63) / 64, 64, 0, stream>>>(h, Wout, bout, out);
}
